// iMeshSegNet_33998961115204
// MI455X (gfx1250) — compile-verified, matrix operand order checked by reading, not run on hardware
//
#include <hip/hip_runtime.h>
#include <stdint.h>

// ---------------- types ----------------
typedef __attribute__((ext_vector_type(16))) _Float16 v16h;
typedef __attribute__((ext_vector_type(8)))  float    v8f;
typedef __attribute__((ext_vector_type(4)))  unsigned int u32x4;
typedef __attribute__((ext_vector_type(8)))  int      i32x8;
typedef __attribute__((ext_vector_type(4)))  int      i32x4;

#define BN_EPS 1e-5f

#if defined(__has_builtin)
#if __has_builtin(__builtin_amdgcn_tensor_load_to_lds) && __has_builtin(__builtin_amdgcn_s_wait_tensorcnt)
#define HAVE_TDM 1
#endif
#endif

// fragment K index for 16-bit WMMA A/B operands (16x32 per ISA layout):
// lanes 0-15 hold K 0-7 (V0-3) and 16-23 (V4-7); lanes 16-31 hold K 8-15 / 24-31.
__device__ __host__ __forceinline__ int klocal(int e, int halfSel) {
  return e + halfSel * 8 + (e >= 8 ? 8 : 0);
}

// ---------------- small utils ----------------
__global__ void fill_zero(float* __restrict__ p, int n) {
  int i = blockIdx.x * blockDim.x + threadIdx.x;
  if (i < n) p[i] = 0.f;
}

// ---------------- weight packing: W(OxC) f32 -> f16 fragments ----------------
__global__ void pack_w(const float* __restrict__ W, _Float16* __restrict__ out,
                       int O, int C, int nRT, int nKT) {
  int tid = blockIdx.x * blockDim.x + threadIdx.x;
  int total = nRT * nKT * 512;
  if (tid >= total) return;
  int e    = tid & 15;
  int lane = (tid >> 4) & 31;
  int t    = tid >> 9;
  int kt   = t % nKT;
  int rt   = t / nKT;
  int row  = rt * 16 + (lane & 15);
  int k    = kt * 32 + klocal(e, lane >> 4);
  float v  = (row < O && k < C) ? W[(size_t)row * C + k] : 0.0f;
  out[tid] = (_Float16)v;
}

#if defined(HAVE_TDM)
// 1-D dword copy global->LDS via Tensor Data Mover (ISA 8.3/8.4 D# layout)
__device__ __forceinline__ void tdm_copy_to_lds(uint32_t lds_off, const void* gptr, uint32_t ndw) {
  uint64_t ga = (uint64_t)gptr;
  u32x4 g0;
  g0[0] = 1u;                                                 // count=1 valid
  g0[1] = lds_off;                                            // lds_addr
  g0[2] = (uint32_t)ga;                                       // global_addr[31:0]
  g0[3] = (uint32_t)((ga >> 32) & 0x1FFFFFFu) | (2u << 30);   // addr[56:32], type=2
  i32x8 g1;
  g1[0] = (int)(2u << 16);                                    // data_size=4B
  g1[1] = (int)((ndw & 0xFFFFu) << 16);                       // tensor_dim0[15:0]
  g1[2] = (int)((ndw >> 16) | (1u << 16));                    // dim0[31:16], tensor_dim1=1
  g1[3] = (int)((ndw & 0xFFFFu) << 16);                       // tile_dim0 = ndw (<=4096)
  g1[4] = 1;                                                  // tile_dim1=1, tile_dim2=0
  g1[5] = (int)ndw;                                           // tensor_dim0_stride
  g1[6] = 0;
  g1[7] = 0;
  i32x4 z4 = {0, 0, 0, 0};
#if __clang_major__ >= 23
  i32x8 z8 = {0, 0, 0, 0, 0, 0, 0, 0};
  __builtin_amdgcn_tensor_load_to_lds(g0, g1, z4, z4, z8, 0);
#else
  __builtin_amdgcn_tensor_load_to_lds(g0, g1, z4, z4, 0);
#endif
}
#endif

// ---------------- WMMA GEMM (specialized on MODE / fused-BN) ----------------
// Out[b, o, r] = sum_c Wpk[o,c] * X(c, b, r) + bias[o]   (bias always valid)
// MODE 0: X planar (B, Cin, Ncols)
// MODE 1: like 0 but applies relu(x*scale[c]+shift[c]) on read (producer BN fused)
// MODE 2: edge-conv reader. Ncols = N*Kn, col -> (n,q). K-tiles are entirely in
//         the x_i half or the (x_j - x_i) half since Cf % 32 == 0 (uniform branch).
template <int MODE, bool BN>
__global__ void __launch_bounds__(128)
gemm_wmma(const _Float16* __restrict__ Wpk, const float* __restrict__ bias,
          const float* __restrict__ X, float* __restrict__ Out,
          const int* __restrict__ idx, const float* __restrict__ scale,
          const float* __restrict__ shift,
          int O, int Cin, int nKT, int Bb, int Ncols, int Npts, int Kn)
{
  __shared__ _Float16 ldsA[8192];                 // up to Cpad=512 (16 K-tiles)
  const int tid  = threadIdx.x;
  const int lane = tid & 31;
  const int rt   = blockIdx.y;
  const int nA   = nKT * 512;

  // stage the packed A row-tile (all K steps) into LDS
#if defined(HAVE_TDM)
  if (tid < 32) {
    tdm_copy_to_lds((uint32_t)(uintptr_t)&ldsA[0], Wpk + (size_t)rt * nA, (uint32_t)(nA >> 1));
    __builtin_amdgcn_s_wait_tensorcnt(0);
  }
  __syncthreads();
#else
  for (int i = tid; i < nA; i += 128) ldsA[i] = Wpk[(size_t)rt * nA + i];
  __syncthreads();
#endif

  const int totalCols = Bb * Ncols;               // max 240000: 32-bit math
  const int colTile = blockIdx.x * 4 + (tid >> 5);
  if (colTile * 16 >= totalCols) return;          // wave-uniform exit (EXEC full at wmma)

  const int cl = lane & 15;
  const int hv = lane >> 4;
  const int colg = colTile * 16 + cl;
  const int col  = colg < totalCols ? colg : totalCols - 1;   // clamp, no divergence
  const int b = col / Ncols;
  const int r = col % Ncols;

  v8f acc = {0.f, 0.f, 0.f, 0.f, 0.f, 0.f, 0.f, 0.f};

  if constexpr (MODE == 2) {
    const int Cf = Cin >> 1;
    const int n = r / Kn, q = r % Kn;
    const int j = idx[((size_t)b * Npts + n) * Kn + q];       // hoisted: 1 load / column
    const float* Xb = X + (size_t)b * Cf * Npts;
    const int nKTh = Cf >> 5;
    for (int kt = 0; kt < nKT; ++kt) {
      v16h afr = *(const v16h*)&ldsA[(kt * 32 + lane) * 16];
      v16h bfr;
      if (kt < nKTh) {                                        // x_i half (uniform)
#pragma unroll
        for (int e = 0; e < 16; ++e) {
          int c = kt * 32 + klocal(e, hv);
          float a = Xb[(size_t)c * Npts + n];
          if constexpr (BN) { a = a * scale[c] + shift[c]; a = a > 0.f ? a : 0.f; }
          bfr[e] = (_Float16)a;
        }
      } else {                                                // x_j - x_i half (uniform)
#pragma unroll
        for (int e = 0; e < 16; ++e) {
          int cj = (kt - nKTh) * 32 + klocal(e, hv);
          const float* pc = Xb + (size_t)cj * Npts;
          float a = pc[n], bj = pc[j];
          if constexpr (BN) {
            float s = scale[cj], h = shift[cj];
            a  = a  * s + h; a  = a  > 0.f ? a  : 0.f;
            bj = bj * s + h; bj = bj > 0.f ? bj : 0.f;
          }
          bfr[e] = (_Float16)(bj - a);
        }
      }
      acc = __builtin_amdgcn_wmma_f32_16x16x32_f16(false, afr, false, bfr,
                                                   (short)0, acc, false, false);
    }
  } else {
    const float* Xb = X + (size_t)b * Cin * Ncols + r;
    for (int kt = 0; kt < nKT; ++kt) {
      v16h afr = *(const v16h*)&ldsA[(kt * 32 + lane) * 16];
      v16h bfr;
#pragma unroll
      for (int e = 0; e < 16; ++e) {
        int c  = kt * 32 + klocal(e, hv);
        int cc = c < Cin ? c : 0;                 // clamped load + select (no exec masking)
        float v = Xb[(size_t)cc * Ncols];
        if constexpr (BN) { v = v * scale[cc] + shift[cc]; v = v > 0.f ? v : 0.f; }
        v = c < Cin ? v : 0.f;
        bfr[e] = (_Float16)v;
      }
      acc = __builtin_amdgcn_wmma_f32_16x16x32_f16(false, afr, false, bfr,
                                                   (short)0, acc, false, false);
    }
  }

  // store epilogue: one base pointer, constant vv*Ncols offsets, hoisted column check
  if (colg < totalCols) {
    const int orow0 = rt * 16 + hv * 8;
    float* po = Out + ((size_t)b * O + orow0) * Ncols + r;
#pragma unroll
    for (int vv = 0; vv < 8; ++vv) {
      if (orow0 + vv < O) po[(size_t)vv * Ncols] = acc[vv] + bias[orow0 + vv];
    }
  }
}

// ---------------- BN statistics: per-channel scale/shift ----------------
__global__ void bn_stats(const float* __restrict__ X, float* __restrict__ scale,
                         float* __restrict__ shift, const float* __restrict__ gamma,
                         const float* __restrict__ beta, int C, int Bb, long Ncols) {
  __shared__ float ssum[256], ssq[256];
  int c = blockIdx.x;
  float s = 0.f, s2 = 0.f;
  for (int b = 0; b < Bb; ++b) {
    const float* p = X + ((size_t)b * C + c) * (size_t)Ncols;
    for (long i = threadIdx.x; i < Ncols; i += 256) {
      float v = p[i]; s += v; s2 += v * v;
    }
  }
  ssum[threadIdx.x] = s; ssq[threadIdx.x] = s2;
  __syncthreads();
  for (int st = 128; st > 0; st >>= 1) {
    if ((int)threadIdx.x < st) {
      ssum[threadIdx.x] += ssum[threadIdx.x + st];
      ssq[threadIdx.x]  += ssq[threadIdx.x + st];
    }
    __syncthreads();
  }
  if (threadIdx.x == 0) {
    float cnt = (float)Bb * (float)Ncols;
    float m   = ssum[0] / cnt;
    float var = ssq[0] / cnt - m * m;
    float sc  = gamma[c] * rsqrtf(var + BN_EPS);
    scale[c] = sc;
    shift[c] = beta[c] - m * sc;
  }
}

// ---------------- max over neighbors (with fused BN+relu) ----------------
__global__ void maxk_bn_relu(const float* __restrict__ X, float* __restrict__ Out,
                             const float* __restrict__ sc, const float* __restrict__ sh,
                             int C, int Npts, int Kn, int ldC, int offC, long total) {
  long i = (long)blockIdx.x * blockDim.x + threadIdx.x;
  if (i >= total) return;
  int n = (int)(i % Npts);
  long t = i / Npts;
  int c = (int)(t % C);
  int b = (int)(t / C);
  const float* p = X + (((size_t)b * C + c) * (size_t)Npts + n) * (size_t)Kn;
  float s = sc[c], h = sh[c], m = -1e30f;
  for (int q = 0; q < Kn; ++q) {
    float v = p[q] * s + h; v = v > 0.f ? v : 0.f;
    if (v > m) m = v;
  }
  Out[((size_t)b * ldC + offC + c) * (size_t)Npts + n] = m;
}

// ---------------- max over points (optional fused BN+relu) ----------------
__global__ void maxn(const float* __restrict__ X, float* __restrict__ Out,
                     const float* __restrict__ sc, const float* __restrict__ sh,
                     int C, int Npts, int ldOut, int offOut, int useBn) {
  __shared__ float sm[256];
  int c = blockIdx.x % C, b = blockIdx.x / C;
  const float* p = X + ((size_t)b * C + c) * (size_t)Npts;
  float s = useBn ? sc[c] : 1.0f, h = useBn ? sh[c] : 0.0f;
  float m = -1e30f;
  for (int i = threadIdx.x; i < Npts; i += 256) {
    float v = p[i];
    if (useBn) { v = v * s + h; v = v > 0.f ? v : 0.f; }
    if (v > m) m = v;
  }
  sm[threadIdx.x] = m;
  __syncthreads();
  for (int st = 128; st > 0; st >>= 1) {
    if ((int)threadIdx.x < st) sm[threadIdx.x] = fmaxf(sm[threadIdx.x], sm[threadIdx.x + st]);
    __syncthreads();
  }
  if (threadIdx.x == 0) Out[(size_t)b * ldOut + offOut + c] = sm[0];
}

// ---------------- kNN: 13 smallest, drop self -> idx6 & idx12 ----------------
#define KNN_TILE 256
__global__ void __launch_bounds__(256)
knn_kernel(const float* __restrict__ pos, int* __restrict__ idxS, int* __restrict__ idxL,
           int Npts) {
  int b = blockIdx.y;
  int i = blockIdx.x * blockDim.x + threadIdx.x;
  const float* px = pos + (size_t)b * 3 * Npts;
  const float* py = px + Npts;
  const float* pz = py + Npts;
  float xi = 0.f, yi = 0.f, zi = 0.f;
  if (i < Npts) { xi = px[i]; yi = py[i]; zi = pz[i]; }
  float bd[13]; int bi[13];
#pragma unroll
  for (int t = 0; t < 13; ++t) { bd[t] = 1e30f; bi[t] = 0; }
  __shared__ float sx[KNN_TILE], sy[KNN_TILE], sz[KNN_TILE];
  for (int j0 = 0; j0 < Npts; j0 += KNN_TILE) {
    int j = j0 + threadIdx.x;
    if (j < Npts) { sx[threadIdx.x] = px[j]; sy[threadIdx.x] = py[j]; sz[threadIdx.x] = pz[j]; }
    if (j0 + KNN_TILE + (int)threadIdx.x < Npts)
      __builtin_prefetch(px + j0 + KNN_TILE + threadIdx.x, 0, 1);   // global_prefetch_b8
    __syncthreads();
    int lim = min(KNN_TILE, Npts - j0);
    if (i < Npts) {
      for (int jj = 0; jj < lim; ++jj) {
        float dx = xi - sx[jj], dy = yi - sy[jj], dz = zi - sz[jj];
        float d = dx * dx + dy * dy + dz * dz;
        if (d < bd[12]) {
          float dn = d; int cn = j0 + jj;
#pragma unroll
          for (int t = 0; t < 13; ++t) {       // register-resident bubble insert
            bool sw = dn < bd[t];
            float td = bd[t]; int ti = bi[t];
            if (sw) { bd[t] = dn; bi[t] = cn; dn = td; cn = ti; }
          }
        }
      }
    }
    __syncthreads();
  }
  if (i < Npts) {
    for (int t = 0; t < 6; ++t)  idxS[((size_t)b * Npts + i) * 6 + t]  = bi[t + 1];
    for (int t = 0; t < 12; ++t) idxL[((size_t)b * Npts + i) * 12 + t] = bi[t + 1];
  }
}

// ---------------- tiny FC layers (B=2) ----------------
__global__ void fc_bn_relu(const float* __restrict__ in, const float* __restrict__ W,
                           const float* __restrict__ bias, const float* __restrict__ gamma,
                           const float* __restrict__ beta, float* __restrict__ out,
                           int K, int O) {
  int o = blockIdx.x * blockDim.x + threadIdx.x;
  if (o >= O) return;
  float h0 = bias[o], h1 = bias[o];
  const float* w = W + (size_t)o * K;
  for (int k = 0; k < K; ++k) { float wv = w[k]; h0 += wv * in[k]; h1 += wv * in[K + k]; }
  float m = 0.5f * (h0 + h1);
  float var = 0.5f * ((h0 - m) * (h0 - m) + (h1 - m) * (h1 - m));
  float sc = gamma[o] * rsqrtf(var + BN_EPS);
  float sh = beta[o] - m * sc;
  float y0 = h0 * sc + sh, y1 = h1 * sc + sh;
  out[o] = y0 > 0.f ? y0 : 0.f;
  out[O + o] = y1 > 0.f ? y1 : 0.f;
}

__global__ void fc_relu(const float* __restrict__ in, const float* __restrict__ W,
                        const float* __restrict__ bias, float* __restrict__ out,
                        int K, int O, int Bb) {
  int o = blockIdx.x * blockDim.x + threadIdx.x;
  if (o >= O) return;
  const float* w = W + (size_t)o * K;
  for (int b = 0; b < Bb; ++b) {
    const float* x = in + (size_t)b * K;
    float h = bias[o];
    for (int k = 0; k < K; ++k) h += w[k] * x[k];
    out[(size_t)b * O + o] = h > 0.f ? h : 0.f;
  }
}

// tT[b][d][c] = (s[b] @ sf3w.T + sf3b + I)[c,d]  (transposed for GEMM weights)
__global__ void make_tT(const float* __restrict__ s, const float* __restrict__ W,
                        const float* __restrict__ bias, float* __restrict__ tT, int K) {
  int id = blockIdx.x * blockDim.x + threadIdx.x;
  if (id >= 2 * 4096) return;
  int b = id / 4096, rc = id % 4096, d = rc / 64, c = rc % 64;
  int row = c * 64 + d;
  const float* w = W + (size_t)row * K;
  const float* x = s + (size_t)b * K;
  float h = bias[row];
  for (int k = 0; k < K; ++k) h += w[k] * x[k];
  if (c == d) h += 1.0f;
  tT[id] = h;
}

// feat = concat([y3 (B,128,N), broadcast g (B,128)], dim=1)
__global__ void build_feat(const float* __restrict__ y3, const float* __restrict__ gf,
                           float* __restrict__ feat, int Npts, long total) {
  long i = (long)blockIdx.x * blockDim.x + threadIdx.x;
  if (i >= total) return;
  int n = (int)(i % Npts);
  long t = i / Npts;
  int c = (int)(t % 256);
  int b = (int)(t / 256);
  feat[i] = (c < 128) ? y3[((size_t)b * 128 + c) * (size_t)Npts + n]
                      : gf[(size_t)b * 128 + (c - 128)];
}

// ---------------- host ----------------
struct Pk { _Float16* w; int nRT, nKT; };
struct SS { float* sc; float* sh; };

static inline int ceilDiv(int a, int b) { return (a + b - 1) / b; }

extern "C" void kernel_launch(void* const* d_in, const int* in_sizes, int n_in,
                              void* d_out, int out_size, void* d_ws, size_t ws_size,
                              hipStream_t stream) {
  (void)in_sizes; (void)n_in; (void)out_size; (void)ws_size;
  const int B = 2, N = 10000, KS = 6, KL = 12;
  const float* x   = (const float*)d_in[0];
  const float* pos = (const float*)d_in[1];
  auto F = [&](int k) { return (const float*)d_in[k]; };

  char* wsb = (char*)d_ws; size_t wso = 0;
  auto alloc = [&](size_t bytes) -> void* {
    void* p = wsb + wso; wso = (wso + bytes + 255) & ~(size_t)255; return p;
  };

  const size_t fN = (size_t)N * 4;
  float* h0   = (float*)alloc((size_t)B * 64 * fN);
  float* h1   = (float*)alloc((size_t)B * 64 * fN);
  float* h2   = (float*)alloc((size_t)B * 64 * fN);
  float* y1   = (float*)alloc((size_t)B * 128 * fN);
  float* y2   = (float*)alloc((size_t)B * 128 * fN);
  float* y3   = (float*)alloc((size_t)B * 128 * fN);
  float* feat = (float*)alloc((size_t)B * 256 * fN);
  float* z1   = (float*)alloc((size_t)B * 256 * fN);
  float* z2   = (float*)alloc((size_t)B * 128 * fN);
  float* S0   = (float*)alloc((size_t)B * 1024 * fN);       // s3 raw / edge tmp2
  float* S1   = (float*)alloc((size_t)B * 64 * 12 * fN);    // s1+s2 / edge tmp1 / m21 raw
  int* idxS   = (int*)alloc((size_t)B * N * KS * 4);
  int* idxL   = (int*)alloc((size_t)B * N * KL * 4);
  float* smax = (float*)alloc((size_t)B * 1024 * 4);
  float* sv1  = (float*)alloc((size_t)B * 512 * 4);
  float* sv2  = (float*)alloc((size_t)B * 256 * 4);
  float* tT   = (float*)alloc((size_t)B * 4096 * 4);
  float* g    = (float*)alloc((size_t)B * 448 * 4);
  float* gg1  = (float*)alloc((size_t)B * 256 * 4);
  float* gg2  = (float*)alloc((size_t)B * 128 * 4);
  float* zbias = (float*)alloc(1024 * 4);                   // zero bias for t-apply
  fill_zero<<<4, 256, 0, stream>>>(zbias, 1024);

  auto packW = [&](const float* W, int O, int C) -> Pk {
    Pk p; p.nRT = ceilDiv(O, 16); p.nKT = ceilDiv(C, 32);
    size_t elems = (size_t)p.nRT * p.nKT * 512;
    p.w = (_Float16*)alloc(elems * 2);
    pack_w<<<ceilDiv((int)elems, 256), 256, 0, stream>>>(W, p.w, O, C, p.nRT, p.nKT);
    return p;
  };
  auto gemm = [&](Pk pk, const float* bias, const float* X, float* Out,
                  const int* idx, const float* sc, const float* sh,
                  int O, int Cin, int Bb, int Ncols, int Kn, int mode) {
    if (!bias) bias = zbias;
    long total = (long)Bb * Ncols;
    dim3 grid((unsigned)((total + 63) / 64), (unsigned)pk.nRT);
    if (mode == 0)
      gemm_wmma<0, false><<<grid, 128, 0, stream>>>(pk.w, bias, X, Out, idx, sc, sh,
                                                    O, Cin, pk.nKT, Bb, Ncols, N, Kn);
    else if (mode == 1)
      gemm_wmma<1, true><<<grid, 128, 0, stream>>>(pk.w, bias, X, Out, idx, sc, sh,
                                                   O, Cin, pk.nKT, Bb, Ncols, N, Kn);
    else if (sc != nullptr)
      gemm_wmma<2, true><<<grid, 128, 0, stream>>>(pk.w, bias, X, Out, idx, sc, sh,
                                                   O, Cin, pk.nKT, Bb, Ncols, N, Kn);
    else
      gemm_wmma<2, false><<<grid, 128, 0, stream>>>(pk.w, bias, X, Out, idx, sc, sh,
                                                    O, Cin, pk.nKT, Bb, Ncols, N, Kn);
  };
  auto bnstats = [&](const float* X, const float* gamma, const float* beta,
                     int C, int Bb, long Ncols) -> SS {
    SS s; s.sc = (float*)alloc((size_t)C * 4); s.sh = (float*)alloc((size_t)C * 4);
    bn_stats<<<C, 256, 0, stream>>>(X, s.sc, s.sh, gamma, beta, C, Bb, Ncols);
    return s;
  };

  // pack all conv weights (param indices follow setup_inputs() dict order)
  Pk pk_m11 = packW(F(2), 64, 15),   pk_m12 = packW(F(6), 64, 64);
  Pk pk_s1  = packW(F(10), 64, 64),  pk_s2  = packW(F(14), 128, 64);
  Pk pk_s3  = packW(F(18), 1024, 128);
  Pk pk_g1s1 = packW(F(32), 64, 128), pk_g1s2 = packW(F(36), 64, 64);
  Pk pk_g1l1 = packW(F(40), 64, 128), pk_g1l2 = packW(F(44), 64, 64);
  Pk pk_m21 = packW(F(48), 128, 128), pk_m22 = packW(F(52), 128, 128);
  Pk pk_g2s1 = packW(F(56), 64, 256), pk_g2s2 = packW(F(60), 64, 64);
  Pk pk_g2l1 = packW(F(64), 64, 256), pk_g2l2 = packW(F(68), 64, 64);
  Pk pk_c1 = packW(F(76), 256, 256),  pk_c2 = packW(F(80), 128, 256);
  Pk pk_c3 = packW(F(84), 66, 128);

  // kNN (both k=6 and k=12 in one pass: keep 13, drop self)
  knn_kernel<<<dim3((unsigned)ceilDiv(N, 256), B), 256, 0, stream>>>(pos, idxS, idxL, N);

  // m11 -> m12 (BN fused into next reader)
  gemm(pk_m11, F(3), x, h0, nullptr, nullptr, nullptr, 64, 15, B, N, 1, 0);
  SS st_m11 = bnstats(h0, F(4), F(5), 64, B, N);
  gemm(pk_m12, F(7), h0, h1, nullptr, st_m11.sc, st_m11.sh, 64, 64, B, N, 1, 1);
  SS st_h1 = bnstats(h1, F(8), F(9), 64, B, N);

  // STNkd(64)
  float* s1o = S1;
  float* s2o = S1 + (size_t)B * 64 * N;
  gemm(pk_s1, F(11), h1, s1o, nullptr, st_h1.sc, st_h1.sh, 64, 64, B, N, 1, 1);
  SS st_s1 = bnstats(s1o, F(12), F(13), 64, B, N);
  gemm(pk_s2, F(15), s1o, s2o, nullptr, st_s1.sc, st_s1.sh, 128, 64, B, N, 1, 1);
  SS st_s2 = bnstats(s2o, F(16), F(17), 128, B, N);
  gemm(pk_s3, F(19), s2o, S0, nullptr, st_s2.sc, st_s2.sh, 1024, 128, B, N, 1, 1);
  SS st_s3 = bnstats(S0, F(20), F(21), 1024, B, N);
  maxn<<<B * 1024, 256, 0, stream>>>(S0, smax, st_s3.sc, st_s3.sh, 1024, N, 1024, 0, 1);
  fc_bn_relu<<<2, 256, 0, stream>>>(smax, F(22), F(23), F(24), F(25), sv1, 1024, 512);
  fc_bn_relu<<<1, 256, 0, stream>>>(sv1, F(26), F(27), F(28), F(29), sv2, 512, 256);
  make_tT<<<ceilDiv(2 * 4096, 256), 256, 0, stream>>>(sv2, F(30), F(31), tT, 256);
  Pk pk_t0 = packW(tT, 64, 64);
  Pk pk_t1 = packW(tT + 4096, 64, 64);
  gemm(pk_t0, nullptr, h1, h2, nullptr, st_h1.sc, st_h1.sh, 64, 64, 1, N, 1, 1);
  gemm(pk_t1, nullptr, h1 + (size_t)64 * N, h2 + (size_t)64 * N,
       nullptr, st_h1.sc, st_h1.sh, 64, 64, 1, N, 1, 1);

  maxn<<<B * 64, 256, 0, stream>>>(h2, g, nullptr, nullptr, 64, N, 448, 0, 0);   // g1

  // edge-conv block: gather fused into GEMM reader, BN fused downstream
  auto edge = [&](const float* src, int Cf, const float* srcSc, const float* srcSh,
                  Pk pkA, const float* bA, const float* gA, const float* eA,
                  Pk pkB, const float* bB, const float* gB, const float* eB,
                  const int* idx, int Kn, float* outBuf, int offC) {
    int NK = N * Kn;
    gemm(pkA, bA, src, S1, idx, srcSc, srcSh, 64, 2 * Cf, B, NK, Kn, 2);
    SS stA = bnstats(S1, gA, eA, 64, B, NK);
    gemm(pkB, bB, S1, S0, nullptr, stA.sc, stA.sh, 64, 64, B, NK, Kn, 1);
    SS stB = bnstats(S0, gB, eB, 64, B, NK);
    long tot = (long)B * 64 * N;
    maxk_bn_relu<<<(unsigned)((tot + 255) / 256), 256, 0, stream>>>(
        S0, outBuf, stB.sc, stB.sh, 64, N, Kn, 128, offC, tot);
  };

  edge(h2, 64, nullptr, nullptr, pk_g1s1, F(33), F(34), F(35),
       pk_g1s2, F(37), F(38), F(39), idxS, KS, y1, 0);
  edge(h2, 64, nullptr, nullptr, pk_g1l1, F(41), F(42), F(43),
       pk_g1l2, F(45), F(46), F(47), idxL, KL, y1, 64);

  maxn<<<B * 128, 256, 0, stream>>>(y1, g, nullptr, nullptr, 128, N, 448, 64, 0); // g2

  gemm(pk_m21, F(49), y1, S1, nullptr, nullptr, nullptr, 128, 128, B, N, 1, 0);
  SS st_m21 = bnstats(S1, F(50), F(51), 128, B, N);
  gemm(pk_m22, F(53), S1, y2, nullptr, st_m21.sc, st_m21.sh, 128, 128, B, N, 1, 1);
  SS st_y2 = bnstats(y2, F(54), F(55), 128, B, N);
  maxn<<<B * 128, 256, 0, stream>>>(y2, g, st_y2.sc, st_y2.sh, 128, N, 448, 192, 1); // g3

  edge(y2, 128, st_y2.sc, st_y2.sh, pk_g2s1, F(57), F(58), F(59),
       pk_g2s2, F(61), F(62), F(63), idxS, KS, y3, 0);
  edge(y2, 128, st_y2.sc, st_y2.sh, pk_g2l1, F(65), F(66), F(67),
       pk_g2l2, F(69), F(70), F(71), idxL, KL, y3, 64);

  maxn<<<B * 128, 256, 0, stream>>>(y3, g, nullptr, nullptr, 128, N, 448, 320, 0); // g4

  fc_relu<<<1, 256, 0, stream>>>(g, F(72), F(73), gg1, 448, 256, B);
  fc_relu<<<1, 256, 0, stream>>>(gg1, F(74), F(75), gg2, 256, 128, B);

  long featTot = (long)B * 256 * N;
  build_feat<<<(unsigned)((featTot + 255) / 256), 256, 0, stream>>>(y3, gg2, feat, N, featTot);

  gemm(pk_c1, F(77), feat, z1, nullptr, nullptr, nullptr, 256, 256, B, N, 1, 0);
  SS st_c1 = bnstats(z1, F(78), F(79), 256, B, N);
  gemm(pk_c2, F(81), z1, z2, nullptr, st_c1.sc, st_c1.sh, 128, 256, B, N, 1, 1);
  SS st_c2 = bnstats(z2, F(82), F(83), 128, B, N);
  gemm(pk_c3, F(85), z2, (float*)d_out, nullptr, st_c2.sc, st_c2.sh, 66, 128, B, N, 1, 1);
}
